// CLIPLoss_14362370638446
// MI455X (gfx1250) — matrix operand order repaired and checked by reading, not compiled, not run on hardware
//
#include <hip/hip_runtime.h>
#include <hip/hip_bf16.h>
#include <math.h>

// ---------------- types for CDNA5 WMMA ----------------
typedef __attribute__((ext_vector_type(16))) __bf16 v16bf;
typedef __attribute__((ext_vector_type(8)))  __bf16 v8bf;
typedef __attribute__((ext_vector_type(8)))  float  v8f;

#define B_ROWS 8192
#define DDIM   1024
#define INV_T  2.0f          // 1 / TEMPERATURE (0.5)
#define NSPLIT 8             // column splits per LSE pass
#define WAVES  8             // waves per workgroup (wave32 => 256 threads)
#define NT     4             // column tiles processed together per wave
#define KCHUNK 128           // K elements staged into LDS per phase
#define COLTILES_TOTAL (B_ROWS / 16)                  // 512
#define COLTILES_PER_SPLIT (COLTILES_TOTAL / NSPLIT)  // 64

// ---------------- 1) row-normalize fp32 -> bf16 ----------------
__global__ __launch_bounds__(256)
void nrm_kernel(const float* __restrict__ src, __hip_bfloat16* __restrict__ dst) {
    __shared__ float red[256];
    const int row = blockIdx.x;
    const int t   = threadIdx.x;
    const float4 v = reinterpret_cast<const float4*>(src + (size_t)row * DDIM)[t];
    red[t] = v.x * v.x + v.y * v.y + v.z * v.z + v.w * v.w;
    __syncthreads();
    for (int off = 128; off > 0; off >>= 1) {
        if (t < off) red[t] += red[t + off];
        __syncthreads();
    }
    const float inv = 1.0f / fmaxf(sqrtf(red[0]), 1e-12f);
    __hip_bfloat16* d = dst + (size_t)row * DDIM + t * 4;
    d[0] = __float2bfloat16(v.x * inv);
    d[1] = __float2bfloat16(v.y * inv);
    d[2] = __float2bfloat16(v.z * inv);
    d[3] = __float2bfloat16(v.w * inv);
}

// ---------------- 2) diagonal dot products (fp32 accum over bf16) ----------------
__global__ __launch_bounds__(256)
void diag_kernel(const __hip_bfloat16* __restrict__ a,
                 const __hip_bfloat16* __restrict__ b,
                 float* __restrict__ dg) {
    const int lane = threadIdx.x & 31;
    const int wave = threadIdx.x >> 5;
    const int row  = blockIdx.x * 8 + wave;
    const __hip_bfloat16* ra = a + (size_t)row * DDIM;
    const __hip_bfloat16* rb = b + (size_t)row * DDIM;
    float acc = 0.f;
    for (int k = lane; k < DDIM; k += 32)
        acc += __bfloat162float(ra[k]) * __bfloat162float(rb[k]);
    #pragma unroll
    for (int m = 16; m >= 1; m >>= 1) acc += __shfl_xor(acc, m, 32);
    if (lane == 0) dg[row] = acc;
}

// ---------------- 3) fused GEMM + online logsumexp ----------------
// Per wave: one 16-row tile of A; NT=4 column tiles per pass. B tiles are
// staged into LDS with CDNA5 async global->LDS copies and shared by all 8
// waves. All NT B fragments are loaded into distinct registers before the
// WMMA burst so the 4 v_wmma issue back-to-back behind one wait.
__global__ __launch_bounds__(256)
void lse_partial_kernel(const __hip_bfloat16* __restrict__ Abf,
                        const __hip_bfloat16* __restrict__ Bbf,
                        float* __restrict__ pm, float* __restrict__ ps) {
    __shared__ __align__(32) __hip_bfloat16 bstage[NT * 16 * KCHUNK];  // 16 KB

    const int tid  = threadIdx.x;
    const int lane = tid & 31;
    const int wave = tid >> 5;
    const int half = lane >> 4;       // 0: lanes 0-15, 1: lanes 16-31
    const int l16  = lane & 15;
    const int mrow = (blockIdx.x * WAVES + wave) * 16;

    // A fragment source: lane holds row M = l16; K elements come from
    // [k0 + half*16 .. +15]  (contiguous, matching the B fragment order)
    const __bf16* arow =
        reinterpret_cast<const __bf16*>(Abf) + (size_t)(mrow + l16) * DDIM + half * 16;

    // Cooperative staging assignment: thread covers 64 contiguous bytes of one
    // (tile,row) 256-byte K-chunk slice.  64 rows x 256 B = 16 KB per phase.
    const int srow = tid >> 2;                     // 0..63 -> tile = srow>>4, r = srow&15
    const int sseg = (tid & 3) * 64;               // byte offset within the row slice
    const unsigned lds_dst0 =
        (unsigned)(size_t)&bstage[0] + (unsigned)(srow * (KCHUNK * 2) + sseg);

    float m[8], s[8];
    #pragma unroll
    for (int r = 0; r < 8; ++r) { m[r] = -INFINITY; s[r] = 0.f; }

    const int jt0 = blockIdx.y * COLTILES_PER_SPLIT;
    for (int g = 0; g < COLTILES_PER_SPLIT; g += NT) {
        v8f c[NT];
        #pragma unroll
        for (int t = 0; t < NT; ++t) c[t] = (v8f){};

        const int jrow = (jt0 + g) * 16;  // first of 64 B-rows in this group
        const __hip_bfloat16* gsrc_row = Bbf + (size_t)(jrow + srow) * DDIM;

        for (int kc = 0; kc < DDIM; kc += KCHUNK) {
            // ---- async stage: 64 rows x KCHUNK bf16 -> LDS (ASYNCcnt path) ----
            {
                unsigned long long gp =
                    (unsigned long long)((const char*)(gsrc_row + kc) + sseg);
                unsigned ld = lds_dst0;
                #pragma unroll
                for (int it = 0; it < 4; ++it) {
                    asm volatile("global_load_async_to_lds_b128 %0, %1, off"
                                 :: "v"(ld), "v"(gp) : "memory");
                    gp += 16; ld += 16;
                }
                asm volatile("s_wait_asynccnt 0x0" ::: "memory");
            }
            __syncthreads();

            // ---- compute: 4 K-steps; gather all fragments, then WMMA burst ----
            #pragma unroll
            for (int kk = 0; kk < KCHUNK; kk += 32) {
                const int k0 = kc + kk;

                // B fragments for all NT tiles into distinct registers:
                // lane holds column N = l16; K elements [kk + half*16 .. +15]
                v16bf b[NT];
                #pragma unroll
                for (int t = 0; t < NT; ++t)
                    b[t] = *reinterpret_cast<const v16bf*>(
                        &bstage[(t * 16 + l16) * KCHUNK + kk + half * 16]);

                v8bf alo = *reinterpret_cast<const v8bf*>(arow + k0);
                v8bf ahi = *reinterpret_cast<const v8bf*>(arow + k0 + 8);
                v16bf a = __builtin_shufflevector(alo, ahi,
                           0,1,2,3,4,5,6,7,8,9,10,11,12,13,14,15);

                // back-to-back WMMAs (disjoint accumulators, shared A)
                #pragma unroll
                for (int t = 0; t < NT; ++t)
                    c[t] = __builtin_amdgcn_wmma_f32_16x16x32_bf16(
                               false, a, false, b[t], (short)0, c[t], false, false);
            }
            __syncthreads();  // protect LDS before next stage overwrites
        }

        // online (max, sum-exp) update; lane's c[t][r] is row (r+8*half), col l16
        #pragma unroll
        for (int t = 0; t < NT; ++t) {
            #pragma unroll
            for (int r = 0; r < 8; ++r) {
                const float v  = INV_T * c[t][r];
                const float mn = fmaxf(m[r], v);
                s[r] = s[r] * __expf(m[r] - mn) + __expf(v - mn);
                m[r] = mn;
            }
        }
    }

    // Merge across the 16 lanes of each half (same rows, different col residues).
    #pragma unroll
    for (int msk = 1; msk < 16; msk <<= 1) {
        #pragma unroll
        for (int r = 0; r < 8; ++r) {
            const float mo = __shfl_xor(m[r], msk, 32);
            const float so = __shfl_xor(s[r], msk, 32);
            const float mn = fmaxf(m[r], mo);
            s[r] = s[r] * __expf(m[r] - mn) + so * __expf(mo - mn);
            m[r] = mn;
        }
    }
    if (l16 == 0) {
        const int rbase = mrow + half * 8;
        float* pmrow = pm + (size_t)blockIdx.y * B_ROWS;
        float* psrow = ps + (size_t)blockIdx.y * B_ROWS;
        #pragma unroll
        for (int r = 0; r < 8; ++r) {
            pmrow[rbase + r] = m[r];
            psrow[rbase + r] = s[r];
        }
    }
}

// ---------------- 4) merge column-split partials into per-row LSE ----------------
__global__ __launch_bounds__(256)
void lse_merge_kernel(const float* __restrict__ pm, const float* __restrict__ ps,
                      float* __restrict__ lse) {
    const int i = blockIdx.x * blockDim.x + threadIdx.x;
    float m = -INFINITY, s = 0.f;
    #pragma unroll
    for (int sp = 0; sp < NSPLIT; ++sp) {
        const float mo = pm[(size_t)sp * B_ROWS + i];
        const float so = ps[(size_t)sp * B_ROWS + i];
        const float mn = fmaxf(m, mo);
        s = s * __expf(m - mn) + so * __expf(mo - mn);
        m = mn;
    }
    lse[i] = m + __logf(s);
}

// ---------------- 5) final loss ----------------
__global__ __launch_bounds__(256)
void loss_kernel(const float* __restrict__ lse_r, const float* __restrict__ lse_c,
                 const float* __restrict__ dg, float* __restrict__ out) {
    __shared__ float red[256];
    float acc = 0.f;
    for (int i = threadIdx.x; i < B_ROWS; i += 256)
        acc += lse_r[i] + lse_c[i] - 4.0f * dg[i];  // sim_diag = 2*dot, once per direction
    red[threadIdx.x] = acc;
    __syncthreads();
    for (int off = 128; off > 0; off >>= 1) {
        if (threadIdx.x < off) red[threadIdx.x] += red[threadIdx.x + off];
        __syncthreads();
    }
    if (threadIdx.x == 0) out[0] = red[0] / (2.0f * (float)B_ROWS);
}

// ---------------- host launcher ----------------
extern "C" void kernel_launch(void* const* d_in, const int* in_sizes, int n_in,
                              void* d_out, int out_size, void* d_ws, size_t ws_size,
                              hipStream_t stream) {
    const float* z_i = (const float*)d_in[0];
    const float* z_j = (const float*)d_in[1];
    float* out = (float*)d_out;

    char* ws = (char*)d_ws;
    size_t off = 0;
    __hip_bfloat16* ziN = (__hip_bfloat16*)(ws + off); off += (size_t)B_ROWS * DDIM * 2;
    __hip_bfloat16* zjN = (__hip_bfloat16*)(ws + off); off += (size_t)B_ROWS * DDIM * 2;
    float* dg    = (float*)(ws + off); off += (size_t)B_ROWS * 4;
    float* pm_r  = (float*)(ws + off); off += (size_t)NSPLIT * B_ROWS * 4;
    float* ps_r  = (float*)(ws + off); off += (size_t)NSPLIT * B_ROWS * 4;
    float* pm_c  = (float*)(ws + off); off += (size_t)NSPLIT * B_ROWS * 4;
    float* ps_c  = (float*)(ws + off); off += (size_t)NSPLIT * B_ROWS * 4;
    float* lse_r = (float*)(ws + off); off += (size_t)B_ROWS * 4;
    float* lse_c = (float*)(ws + off); off += (size_t)B_ROWS * 4;

    // 1) normalize + bf16 convert
    nrm_kernel<<<B_ROWS, 256, 0, stream>>>(z_i, ziN);
    nrm_kernel<<<B_ROWS, 256, 0, stream>>>(z_j, zjN);

    // 2) diagonal dots
    diag_kernel<<<B_ROWS / 8, 256, 0, stream>>>(ziN, zjN, dg);

    // 3) fused GEMM + online LSE, both directions
    dim3 grid(B_ROWS / (16 * WAVES), NSPLIT);           // (64, 8)
    lse_partial_kernel<<<grid, 256, 0, stream>>>(ziN, zjN, pm_r, ps_r); // rows of sim
    lse_partial_kernel<<<grid, 256, 0, stream>>>(zjN, ziN, pm_c, ps_c); // rows of sim^T

    // 4) merge partials
    lse_merge_kernel<<<B_ROWS / 256, 256, 0, stream>>>(pm_r, ps_r, lse_r);
    lse_merge_kernel<<<B_ROWS / 256, 256, 0, stream>>>(pm_c, ps_c, lse_c);

    // 5) final reduction
    loss_kernel<<<1, 256, 0, stream>>>(lse_r, lse_c, dg, out);
}
